// NMT_41979010351815
// MI455X (gfx1250) — matrix operand order repaired and checked by reading, not compiled, not run on hardware
//
#include <hip/hip_runtime.h>
#include <hip/hip_bf16.h>

typedef __attribute__((ext_vector_type(16))) __bf16 v16bf;
typedef __attribute__((ext_vector_type(8)))  __bf16 v8bf;
typedef __attribute__((ext_vector_type(8)))  float  v8f;

#define SLEN 40
#define TLEN 40
#define BATCH 64
#define NROWS (SLEN * BATCH)          // 2560
#define EMB 300
#define EMBP 320
#define ENC_H 512
#define DEC_H 1024
#define VOCAB 23262
#define VOCABP 23264

#define MT 2                          // 16-row tiles per wave
#define NT 4                          // 16-col tiles per wave

__device__ __forceinline__ __bf16 f2bf(float x) {
  unsigned u = __builtin_bit_cast(unsigned, x);
  unsigned r = u + 0x7FFFu + ((u >> 16) & 1u);   // round-to-nearest-even
  unsigned short h = (unsigned short)(r >> 16);
  return __builtin_bit_cast(__bf16, h);
}

__device__ __forceinline__ float sigmf(float x) { return 1.f / (1.f + __expf(-x)); }

// ---------------------------------------------------------------------------
// WMMA GEMM: Cout[M,N] = act( Cin?[m,n] + sum_k A[m,k]*W[n,k] + bias?[n] )
// A: bf16 [M,K] row-major, W: bf16 [N,K] row-major, K % 32 == 0, M % 32 == 0.
// One wave computes a 32x64 macro-tile (2M x 4N of 16x16); 4 waves / block.
// Optional bf16 mirror of the output in Cbf (for chained GEMMs).
// ---------------------------------------------------------------------------
__global__ void __launch_bounds__(128)
nmt_wmma_gemm(const __bf16* __restrict__ A, const __bf16* __restrict__ W,
              const float* __restrict__ Cin, const float* __restrict__ bias,
              float* __restrict__ Cout, __bf16* __restrict__ Cbf,
              int M, int N, int K, int act)
{
  const int lane = threadIdx.x & 31;
  const int wave = threadIdx.x >> 5;
  const int n0   = (blockIdx.x * 4 + wave) * NT;  // first N tile of this wave
  const int m0   = blockIdx.y * MT;               // first M tile of this wave
  if (n0 * 16 >= N) return;                       // wave-uniform (never true w/ exact grid)

  const int row = lane & 15;
  const int hi  = lane >> 4;
  const int ka  = hi * 8;    // A 16-bit layout: lanes 16-31 hold K+8 / K+24 runs
  const int kb  = hi * 8;    // B 16-bit layout: lanes 16-31 hold K+8 / K+24 runs

  const __bf16* Ar[MT];
#pragma unroll
  for (int mt = 0; mt < MT; ++mt)
    Ar[mt] = A + (size_t)((m0 + mt) * 16 + row) * K;

  bool nok[NT];
  const __bf16* Wr[NT];
#pragma unroll
  for (int nt = 0; nt < NT; ++nt) {
    nok[nt] = (n0 + nt) * 16 < N;
    Wr[nt]  = W + (size_t)(((nok[nt] ? n0 + nt : n0) * 16) + row) * K;  // safe redirect
  }

  v8f zero = {};
  v8f acc[MT][NT];
#pragma unroll
  for (int mt = 0; mt < MT; ++mt)
#pragma unroll
    for (int nt = 0; nt < NT; ++nt) acc[mt][nt] = zero;

  for (int k0 = 0; k0 < K; k0 += 32) {
    v16bf a[MT];
#pragma unroll
    for (int mt = 0; mt < MT; ++mt) {
      v8bf alo = *(const v8bf*)(Ar[mt] + k0 + ka);        // K run 0..7
      v8bf ahi = *(const v8bf*)(Ar[mt] + k0 + ka + 16);   // K run 16..23
      a[mt] = __builtin_shufflevector(alo, ahi, 0, 1, 2, 3, 4, 5, 6, 7,
                                      8, 9, 10, 11, 12, 13, 14, 15);
    }
#pragma unroll
    for (int nt = 0; nt < NT; ++nt) {
      v8bf blo = *(const v8bf*)(Wr[nt] + k0 + kb);        // K run 0..7
      v8bf bhi = *(const v8bf*)(Wr[nt] + k0 + kb + 16);   // K run 16..23
      v16bf b = __builtin_shufflevector(blo, bhi, 0, 1, 2, 3, 4, 5, 6, 7,
                                        8, 9, 10, 11, 12, 13, 14, 15);
#pragma unroll
      for (int mt = 0; mt < MT; ++mt)
        acc[mt][nt] = __builtin_amdgcn_wmma_f32_16x16x32_bf16(
            false, a[mt], false, b, (short)0, acc[mt][nt], false, false);
    }
  }

#pragma unroll
  for (int nt = 0; nt < NT; ++nt) {
    if (!nok[nt]) continue;
    const int cn = (n0 + nt) * 16 + row;
    const float bv = bias ? bias[cn] : 0.f;
#pragma unroll
    for (int mt = 0; mt < MT; ++mt) {
      const int cm = (m0 + mt) * 16 + hi * 8;
#pragma unroll
      for (int r = 0; r < 8; ++r) {
        size_t idx = (size_t)(cm + r) * N + cn;
        float v = acc[mt][nt][r] + bv;
        if (Cin) v += Cin[idx];
        if (act) v = tanhf(v);
        Cout[idx] = v;
        if (Cbf) Cbf[idx] = f2bf(v);
      }
    }
  }
}

// ---------------------------------------------------------------------------
// Weight conversion / elementwise kernels
// ---------------------------------------------------------------------------
__global__ void nmt_cvt_pad(const float* __restrict__ W, __bf16* __restrict__ out,
                            int Nout, int Nsrc, int K, int Kp, int srcld, int srcoff)
{
  size_t i = (size_t)blockIdx.x * blockDim.x + threadIdx.x;
  if (i >= (size_t)Nout * Kp) return;
  int n = (int)(i / Kp), k = (int)(i % Kp);
  float v = (n < Nsrc && k < K) ? W[(size_t)n * srcld + srcoff + k] : 0.f;
  out[i] = f2bf(v);
}

__global__ void nmt_cvt_T(const float* __restrict__ W, __bf16* __restrict__ out, int N)
{
  size_t i = (size_t)blockIdx.x * blockDim.x + threadIdx.x;
  if (i >= (size_t)N * N) return;
  int n = (int)(i / N), k = (int)(i % N);
  out[i] = f2bf(W[(size_t)k * N + n]);      // out[n][k] = W[k][n]  (enc_out @ W_i)
}

__global__ void nmt_cvt_act(const float* __restrict__ in, __bf16* __restrict__ out, int n)
{
  int i = blockIdx.x * blockDim.x + threadIdx.x;
  if (i < n) out[i] = f2bf(in[i]);
}

__global__ void nmt_vecadd(const float* __restrict__ a, const float* __restrict__ b,
                           float* __restrict__ o, int n)
{
  int i = blockIdx.x * blockDim.x + threadIdx.x;
  if (i < n) o[i] = a[i] + b[i];
}

// embedding gather straight to padded bf16 rows
__global__ void nmt_gather_bf(const int* __restrict__ idx, const float* __restrict__ table,
                              __bf16* __restrict__ out, int nrows, int wsrc, int wdst)
{
  size_t i = (size_t)blockIdx.x * blockDim.x + threadIdx.x;
  if (i >= (size_t)nrows * wdst) return;
  int r = (int)(i / wdst), c = (int)(i % wdst);
  out[i] = f2bf(c < wsrc ? table[(size_t)idx[r] * wsrc + c] : 0.f);
}

// gates layout [B, 4H] in PyTorch order i,f,g,o; writes f32 + bf16 state copies
__global__ void nmt_lstm_cell(const float* __restrict__ gates, float* __restrict__ c,
                              float* __restrict__ h, __bf16* __restrict__ hbf,
                              float* __restrict__ hout, __bf16* __restrict__ houtbf,
                              int hout_ld, int H)
{
  int i = blockIdx.x * blockDim.x + threadIdx.x;
  if (i >= BATCH * H) return;
  int b = i / H, j = i - b * H;
  const float* g = gates + (size_t)b * 4 * H;
  float ig = sigmf(g[j]);
  float fg = sigmf(g[H + j]);
  float gg = tanhf(g[2 * H + j]);
  float og = sigmf(g[3 * H + j]);
  float cn = fg * c[i] + ig * gg;
  float hn = og * tanhf(cn);
  c[i] = cn;
  h[i] = hn;
  hbf[i] = f2bf(hn);
  if (hout)   hout[(size_t)b * hout_ld + j] = hn;
  if (houtbf) houtbf[(size_t)b * hout_ld + j] = f2bf(hn);
}

// scores[b][s] = dot(mat_left[s][b][:], d_h[b][:])
__global__ void nmt_attn_scores(const float* __restrict__ mat_left,
                                const float* __restrict__ dh, float* __restrict__ scores)
{
  int i = blockIdx.x * blockDim.x + threadIdx.x;
  if (i >= BATCH * SLEN) return;
  int b = i / SLEN, s = i - b * SLEN;
  const float* ml = mat_left + ((size_t)s * BATCH + b) * DEC_H;
  const float* hh = dh + (size_t)b * DEC_H;
  float acc = 0.f;
  for (int d = 0; d < DEC_H; ++d) acc += ml[d] * hh[d];
  scores[i] = acc;
}

__global__ void nmt_softmax40(float* __restrict__ scores)
{
  int b = blockIdx.x * blockDim.x + threadIdx.x;
  if (b >= BATCH) return;
  float* s = scores + b * SLEN;
  float mx = -3.4e38f;
  for (int i = 0; i < SLEN; ++i) mx = fmaxf(mx, s[i]);
  float sum = 0.f;
  for (int i = 0; i < SLEN; ++i) { float e = __expf(s[i] - mx); s[i] = e; sum += e; }
  float inv = 1.f / sum;
  for (int i = 0; i < SLEN; ++i) s[i] *= inv;
}

// s_t[b][d] = sum_s attn[b][s] * enc_out[s][b][d]  (bf16 result feeds W_o GEMM)
__global__ void nmt_attn_apply(const float* __restrict__ attn,
                               const float* __restrict__ enc_out, __bf16* __restrict__ st)
{
  int i = blockIdx.x * blockDim.x + threadIdx.x;
  if (i >= BATCH * DEC_H) return;
  int b = i >> 10, d = i & 1023;
  float acc = 0.f;
  for (int s = 0; s < SLEN; ++s)
    acc += attn[b * SLEN + s] * enc_out[((size_t)s * BATCH + b) * DEC_H + d];
  st[i] = f2bf(acc);
}

// one block per row: out[b][v] = x[b][v]+bias[v] - logsumexp
__global__ void nmt_log_softmax(const float* __restrict__ logits,
                                const float* __restrict__ bias, float* __restrict__ out,
                                int V, int ldl)
{
  __shared__ float red[256];
  int b = blockIdx.x;
  const float* x = logits + (size_t)b * ldl;
  float* o = out + (size_t)b * V;
  float mx = -3.4e38f;
  for (int i = threadIdx.x; i < V; i += blockDim.x) mx = fmaxf(mx, x[i] + bias[i]);
  red[threadIdx.x] = mx; __syncthreads();
  for (int off = 128; off > 0; off >>= 1) {
    if (threadIdx.x < off) red[threadIdx.x] = fmaxf(red[threadIdx.x], red[threadIdx.x + off]);
    __syncthreads();
  }
  mx = red[0]; __syncthreads();
  float sum = 0.f;
  for (int i = threadIdx.x; i < V; i += blockDim.x) sum += __expf(x[i] + bias[i] - mx);
  red[threadIdx.x] = sum; __syncthreads();
  for (int off = 128; off > 0; off >>= 1) {
    if (threadIdx.x < off) red[threadIdx.x] += red[threadIdx.x + off];
    __syncthreads();
  }
  float lse = __logf(red[0]) + mx;
  for (int i = threadIdx.x; i < V; i += blockDim.x) o[i] = x[i] + bias[i] - lse;
}

// ---------------------------------------------------------------------------
// Host orchestration
// ---------------------------------------------------------------------------
static inline void launch_gemm(hipStream_t s, const __bf16* A, const __bf16* W,
                               const float* Cin, const float* bias, float* C, __bf16* Cbf,
                               int M, int N, int K, int act)
{
  dim3 grid((N / 16 + 4 * NT - 1) / (4 * NT), M / (16 * MT));
  nmt_wmma_gemm<<<grid, dim3(128), 0, s>>>(A, W, Cin, bias, C, Cbf, M, N, K, act);
}

static inline int blk(long n) { return (int)((n + 255) / 256); }

extern "C" void kernel_launch(void* const* d_in, const int* in_sizes, int n_in,
                              void* d_out, int out_size, void* d_ws, size_t ws_size,
                              hipStream_t stream)
{
  const int*   src_idx = (const int*)d_in[0];
  const int*   trg_idx = (const int*)d_in[1];
  const float* d_h0    = (const float*)d_in[2];
  const float* d_c0    = (const float*)d_in[3];
  const float* emb_en  = (const float*)d_in[4];
  const float* emb_de  = (const float*)d_in[5];
  const float* W_ih_f  = (const float*)d_in[6];
  const float* W_hh_f  = (const float*)d_in[7];
  const float* b_ih_f  = (const float*)d_in[8];
  const float* b_hh_f  = (const float*)d_in[9];
  const float* W_ih_r  = (const float*)d_in[10];
  const float* W_hh_r  = (const float*)d_in[11];
  const float* b_ih_r  = (const float*)d_in[12];
  const float* b_hh_r  = (const float*)d_in[13];
  const float* W_ih_de = (const float*)d_in[14];
  const float* W_hh_de = (const float*)d_in[15];
  const float* b_ih_de = (const float*)d_in[16];
  const float* b_hh_de = (const float*)d_in[17];
  const float* W_i     = (const float*)d_in[18];
  const float* W_o     = (const float*)d_in[19];
  const float* W_gen   = (const float*)d_in[20];
  const float* b_gen   = (const float*)d_in[21];
  float* out = (float*)d_out;

  // ---- workspace layout ----
  char* ws = (char*)d_ws;
  size_t off = 0;
  auto alloc = [&](size_t elems, size_t esz) -> char* {
    char* p = ws + off;
    off += (elems * esz + 255) & ~(size_t)255;
    return p;
  };
  // bf16 weights (K padded to %32, generator rows padded to %16)
  __bf16* wihf  = (__bf16*)alloc((size_t)2048 * EMBP, 2);
  __bf16* wihr  = (__bf16*)alloc((size_t)2048 * EMBP, 2);
  __bf16* whhf  = (__bf16*)alloc((size_t)2048 * ENC_H, 2);
  __bf16* whhr  = (__bf16*)alloc((size_t)2048 * ENC_H, 2);
  __bf16* wdec  = (__bf16*)alloc((size_t)4096 * 1024, 2);   // W_ih_de[:, :1024]
  __bf16* wdee  = (__bf16*)alloc((size_t)4096 * EMBP, 2);   // W_ih_de[:, 1024:1324]
  __bf16* whhde = (__bf16*)alloc((size_t)4096 * 1024, 2);
  __bf16* wiT   = (__bf16*)alloc((size_t)1024 * 1024, 2);   // W_i transposed
  __bf16* wo1   = (__bf16*)alloc((size_t)1024 * 1024, 2);   // W_o[:, :1024]
  __bf16* wo2   = (__bf16*)alloc((size_t)1024 * 1024, 2);   // W_o[:, 1024:2048]
  __bf16* wgen  = (__bf16*)alloc((size_t)VOCABP * 1024, 2);
  // bf16 activations
  __bf16* xsrcb   = (__bf16*)alloc((size_t)NROWS * EMBP, 2);
  __bf16* dembb   = (__bf16*)alloc((size_t)NROWS * EMBP, 2);
  __bf16* encoutb = (__bf16*)alloc((size_t)NROWS * 1024, 2);
  __bf16* hencb   = (__bf16*)alloc((size_t)BATCH * ENC_H, 2);
  __bf16* dhb     = (__bf16*)alloc((size_t)BATCH * DEC_H, 2);
  __bf16* ctb     = (__bf16*)alloc((size_t)BATCH * DEC_H, 2);
  __bf16* stb     = (__bf16*)alloc((size_t)BATCH * DEC_H, 2);
  // f32 activations
  float* gxf     = (float*)alloc((size_t)NROWS * 2048, 4);
  float* gxr     = (float*)alloc((size_t)NROWS * 2048, 4);
  float* encout  = (float*)alloc((size_t)NROWS * 1024, 4);
  float* matleft = (float*)alloc((size_t)NROWS * 1024, 4);
  float* deprex  = (float*)alloc((size_t)NROWS * 4096, 4);
  float* bencf   = (float*)alloc(2048, 4);
  float* bencr   = (float*)alloc(2048, 4);
  float* bde     = (float*)alloc(4096, 4);
  float* hf      = (float*)alloc((size_t)BATCH * ENC_H, 4);
  float* cf      = (float*)alloc((size_t)BATCH * ENC_H, 4);
  float* hr      = (float*)alloc((size_t)BATCH * ENC_H, 4);
  float* cr      = (float*)alloc((size_t)BATCH * ENC_H, 4);
  float* dh      = (float*)alloc((size_t)BATCH * DEC_H, 4);
  float* dc      = (float*)alloc((size_t)BATCH * DEC_H, 4);
  float* gates   = (float*)alloc((size_t)BATCH * 4096, 4);
  float* scores  = (float*)alloc((size_t)BATCH * SLEN, 4);
  float* tmp1    = (float*)alloc((size_t)BATCH * DEC_H, 4);
  float* ct      = (float*)alloc((size_t)BATCH * DEC_H, 4);
  float* logits  = (float*)alloc((size_t)BATCH * VOCABP, 4);

  // ---- 1. weight conversion to padded bf16 ----
  nmt_cvt_pad<<<blk((long)2048 * EMBP), 256, 0, stream>>>(W_ih_f, wihf, 2048, 2048, EMB, EMBP, EMB, 0);
  nmt_cvt_pad<<<blk((long)2048 * EMBP), 256, 0, stream>>>(W_ih_r, wihr, 2048, 2048, EMB, EMBP, EMB, 0);
  nmt_cvt_pad<<<blk((long)2048 * ENC_H), 256, 0, stream>>>(W_hh_f, whhf, 2048, 2048, ENC_H, ENC_H, ENC_H, 0);
  nmt_cvt_pad<<<blk((long)2048 * ENC_H), 256, 0, stream>>>(W_hh_r, whhr, 2048, 2048, ENC_H, ENC_H, ENC_H, 0);
  nmt_cvt_pad<<<blk((long)4096 * 1024), 256, 0, stream>>>(W_ih_de, wdec, 4096, 4096, 1024, 1024, 1324, 0);
  nmt_cvt_pad<<<blk((long)4096 * EMBP), 256, 0, stream>>>(W_ih_de, wdee, 4096, 4096, EMB, EMBP, 1324, 1024);
  nmt_cvt_pad<<<blk((long)4096 * 1024), 256, 0, stream>>>(W_hh_de, whhde, 4096, 4096, 1024, 1024, 1024, 0);
  nmt_cvt_T<<<blk((long)1024 * 1024), 256, 0, stream>>>(W_i, wiT, 1024);
  nmt_cvt_pad<<<blk((long)1024 * 1024), 256, 0, stream>>>(W_o, wo1, 1024, 1024, 1024, 1024, 2048, 0);
  nmt_cvt_pad<<<blk((long)1024 * 1024), 256, 0, stream>>>(W_o, wo2, 1024, 1024, 1024, 1024, 2048, 1024);
  nmt_cvt_pad<<<blk((long)VOCABP * 1024), 256, 0, stream>>>(W_gen, wgen, VOCABP, VOCAB, 1024, 1024, 1024, 0);
  nmt_vecadd<<<blk(2048), 256, 0, stream>>>(b_ih_f, b_hh_f, bencf, 2048);
  nmt_vecadd<<<blk(2048), 256, 0, stream>>>(b_ih_r, b_hh_r, bencr, 2048);
  nmt_vecadd<<<blk(4096), 256, 0, stream>>>(b_ih_de, b_hh_de, bde, 4096);

  // ---- 2. embedding gathers (padded bf16) ----
  nmt_gather_bf<<<blk((long)NROWS * EMBP), 256, 0, stream>>>(src_idx, emb_en, xsrcb, NROWS, EMB, EMBP);
  nmt_gather_bf<<<blk((long)NROWS * EMBP), 256, 0, stream>>>(trg_idx, emb_de, dembb, NROWS, EMB, EMBP);

  // ---- 3. batched input projections (hoisted out of recurrences) ----
  launch_gemm(stream, xsrcb, wihf, nullptr, bencf, gxf, nullptr, NROWS, 2048, EMBP, 0);
  launch_gemm(stream, xsrcb, wihr, nullptr, bencr, gxr, nullptr, NROWS, 2048, EMBP, 0);
  launch_gemm(stream, dembb, wdee, nullptr, bde, deprex, nullptr, NROWS, 4096, EMBP, 0);

  // ---- 4. bidirectional encoder ----
  hipMemsetAsync(hf, 0, (size_t)BATCH * ENC_H * 4, stream);
  hipMemsetAsync(cf, 0, (size_t)BATCH * ENC_H * 4, stream);
  hipMemsetAsync(hencb, 0, (size_t)BATCH * ENC_H * 2, stream);
  for (int t = 0; t < SLEN; ++t) {
    launch_gemm(stream, hencb, whhf, gxf + (size_t)t * BATCH * 2048, nullptr, gates, nullptr,
                BATCH, 2048, ENC_H, 0);
    nmt_lstm_cell<<<blk(BATCH * ENC_H), 256, 0, stream>>>(
        gates, cf, hf, hencb,
        encout + (size_t)t * BATCH * 1024, encoutb + (size_t)t * BATCH * 1024, 1024, ENC_H);
  }
  hipMemsetAsync(hr, 0, (size_t)BATCH * ENC_H * 4, stream);
  hipMemsetAsync(cr, 0, (size_t)BATCH * ENC_H * 4, stream);
  hipMemsetAsync(hencb, 0, (size_t)BATCH * ENC_H * 2, stream);
  for (int t = SLEN - 1; t >= 0; --t) {
    launch_gemm(stream, hencb, whhr, gxr + (size_t)t * BATCH * 2048, nullptr, gates, nullptr,
                BATCH, 2048, ENC_H, 0);
    nmt_lstm_cell<<<blk(BATCH * ENC_H), 256, 0, stream>>>(
        gates, cr, hr, hencb,
        encout + (size_t)t * BATCH * 1024 + ENC_H,
        encoutb + (size_t)t * BATCH * 1024 + ENC_H, 1024, ENC_H);
  }

  // ---- 5. mat_left = enc_out @ W_i ----
  launch_gemm(stream, encoutb, wiT, nullptr, nullptr, matleft, nullptr, NROWS, 1024, 1024, 0);

  // ---- 6. attention decoder ----
  hipMemcpyAsync(dh, d_h0, (size_t)BATCH * DEC_H * 4, hipMemcpyDeviceToDevice, stream);
  hipMemcpyAsync(dc, d_c0, (size_t)BATCH * DEC_H * 4, hipMemcpyDeviceToDevice, stream);
  nmt_cvt_act<<<blk(BATCH * DEC_H), 256, 0, stream>>>(d_h0, dhb, BATCH * DEC_H);
  for (int t = 0; t < TLEN; ++t) {
    nmt_attn_scores<<<blk(BATCH * SLEN), 256, 0, stream>>>(matleft, dh, scores);
    nmt_softmax40<<<1, 64, 0, stream>>>(scores);
    nmt_attn_apply<<<blk(BATCH * DEC_H), 256, 0, stream>>>(scores, encout, stb);
    // c_t = tanh([s_t, d_h] @ W_o^T)
    launch_gemm(stream, stb, wo1, nullptr, nullptr, tmp1, nullptr, BATCH, 1024, 1024, 0);
    launch_gemm(stream, dhb, wo2, tmp1, nullptr, ct, ctb, BATCH, 1024, 1024, 1);
    // gates = deprex[t] + c_t @ Wde_c^T + d_h @ W_hh_de^T  (biases folded into deprex)
    launch_gemm(stream, ctb, wdec, deprex + (size_t)t * BATCH * 4096, nullptr, gates, nullptr,
                BATCH, 4096, 1024, 0);
    launch_gemm(stream, dhb, whhde, gates, nullptr, gates, nullptr, BATCH, 4096, 1024, 0);
    nmt_lstm_cell<<<blk(BATCH * DEC_H), 256, 0, stream>>>(gates, dc, dh, dhb,
                                                          nullptr, nullptr, 0, DEC_H);
    // generator + log-softmax
    launch_gemm(stream, dhb, wgen, nullptr, nullptr, logits, nullptr, BATCH, VOCABP, 1024, 0);
    nmt_log_softmax<<<BATCH, 256, 0, stream>>>(logits, b_gen,
                                               out + (size_t)t * BATCH * VOCAB, VOCAB, VOCABP);
  }
}
